// Transformer_61375082660563
// MI455X (gfx1250) — matrix operand order repaired and checked by reading, not compiled, not run on hardware
//
#include <hip/hip_runtime.h>

// ---------------------------------------------------------------------------
// CDNA5 (gfx1250) bf16 WMMA transformer forward (single + pairwise tracks)
// ---------------------------------------------------------------------------

typedef __attribute__((ext_vector_type(16))) __bf16 v16bf;
typedef __attribute__((ext_vector_type(8)))  float  v8f;
typedef __attribute__((ext_vector_type(4)))  unsigned int v4u;

union Frag { v16bf v; v4u q[2]; unsigned int u[8]; };

// Single-instruction pack of two f32 into packed bf16x2.
__device__ __forceinline__ unsigned int pack2(float a, float b) {
    unsigned int r;
    asm("v_cvt_pk_bf16_f32 %0, %1, %2" : "=v"(r) : "v"(a), "v"(b));
    return r;
}

// A-matrix fragment (16xK tile rows): lane holds row = lane%16, K swizzled
// per ISA 7.12.2 (16-bit A 16x32): element e -> K = (e&8)*2 + 8*half + (e&7).
// Two contiguous 16B runs -> 2x ds_load_b128 (row stride must be 16B-aligned).
__device__ __forceinline__ v16bf frag_a(const unsigned int* row, int kpb) {
    const int half = (threadIdx.x >> 4) & 1;
    Frag f;
    f.q[0] = *(const v4u*)(row + kpb + half * 4);
    f.q[1] = *(const v4u*)(row + kpb + 8 + half * 4);
    return f.v;
}

// B-matrix fragment (Kx16 tile): lane holds col = lane%16; element e ->
// K = (e&8)*2 + 8*half + (e&7). Two 16B runs -> 2x ds_load_b128.
__device__ __forceinline__ v16bf frag_b(const unsigned int* row, int kpb) {
    const int half = (threadIdx.x >> 4) & 1;
    const unsigned int* p = row + kpb + half * 4;
    Frag f;
    f.q[0] = *(const v4u*)(p);
    f.q[1] = *(const v4u*)(p + 8);
    return f.v;
}

__device__ __forceinline__ v8f wmma_bf16(v16bf a, v16bf b, v8f c) {
    return __builtin_amdgcn_wmma_f32_16x16x32_bf16(
        false, a, false, b, (short)0, c, false, false);
}

// ---------------------------------------------------------------------------
// Generic bf16-compute GEMM: C[M,N] = A[M,K] * B[K,N] (+bias) (+res) (+relu)
// block = 256 threads (8 waves), tile 128x128, K-step 32, double-buffered LDS.
// Row stride 20 uints (80B): 16B-aligned rows, bank-group stride 5 (coprime
// with 16) -> conflict-free ds_load_b128 across a 16-lane half.
// Requires M % 128 == 0 and K % 32 == 0 (true for all call sites).
// ---------------------------------------------------------------------------
__global__ __launch_bounds__(256) void k_gemm(
    const float* __restrict__ A, const float* __restrict__ B,
    const float* __restrict__ bias, const float* __restrict__ res,
    float* __restrict__ C, int M, int N, int K, int relu)
{
    __shared__ __attribute__((aligned(16))) unsigned int Als[2][128][20];
    __shared__ __attribute__((aligned(16))) unsigned int Bls[2][128][20];
    const int tid = threadIdx.x;
    const int w = tid >> 5, lane = tid & 31, half = lane >> 4, ln = lane & 15;
    const int wm = w >> 1, wn = w & 1;
    const int bm = blockIdx.y * 128, bn = blockIdx.x * 128;

    // Per-thread staging coordinates (8 chunks each for A and B).
    const int am = tid >> 4, akp = tid & 15;          // +16 rows per chunk
    const int bkp = tid >> 7, bnn = tid & 127;        // +2 kpairs per chunk
    const int gnc = (bn + bnn < N) ? (bn + bnn) : (N - 1);
    const bool nok = (bn + bnn) < N;

    float a0[8], a1[8], b0[8], b1[8];

    auto load_tile = [&](int k0) {
#pragma unroll
        for (int it = 0; it < 8; ++it) {
            const float* p = A + (size_t)(bm + am + 16 * it) * K + k0 + 2 * akp;
            a0[it] = p[0]; a1[it] = p[1];
        }
#pragma unroll
        for (int it = 0; it < 8; ++it) {
            const float* p = B + (size_t)(k0 + 2 * (bkp + 2 * it)) * N + gnc;
            float x0 = p[0], x1 = p[N];
            b0[it] = nok ? x0 : 0.f;
            b1[it] = nok ? x1 : 0.f;
        }
    };
    auto store_tile = [&](int buf) {
#pragma unroll
        for (int it = 0; it < 8; ++it)
            Als[buf][am + 16 * it][akp] = pack2(a0[it], a1[it]);
#pragma unroll
        for (int it = 0; it < 8; ++it)
            Bls[buf][bnn][bkp + 2 * it] = pack2(b0[it], b1[it]);
    };

    const v8f vz = {0.f, 0.f, 0.f, 0.f, 0.f, 0.f, 0.f, 0.f};
    v8f acc[2][4];
#pragma unroll
    for (int a_ = 0; a_ < 2; ++a_)
#pragma unroll
        for (int b_ = 0; b_ < 4; ++b_) acc[a_][b_] = vz;

    const int nk = K >> 5;
    load_tile(0);
    store_tile(0);
    __syncthreads();

    for (int kt = 0; kt < nk; ++kt) {
        const int cur = kt & 1;
        if (kt + 1 < nk) load_tile((kt + 1) << 5);     // overlap with WMMA burst
        if (kt + 2 < nk) {                             // pull tile kt+2 toward L2
            __builtin_prefetch(A + (size_t)(bm + am) * K + ((kt + 2) << 5) + 2 * akp, 0, 0);
            __builtin_prefetch(B + (size_t)(((kt + 2) << 5) + 2 * bkp) * N + gnc, 0, 0);
        }

        v16bf af0 = frag_a(&Als[cur][wm * 32 + ln][0], 0);
        v16bf af1 = frag_a(&Als[cur][wm * 32 + 16 + ln][0], 0);
#pragma unroll
        for (int ns = 0; ns < 4; ++ns) {
            v16bf bf = frag_b(&Bls[cur][wn * 64 + ns * 16 + ln][0], 0);
            acc[0][ns] = wmma_bf16(af0, bf, acc[0][ns]);
            acc[1][ns] = wmma_bf16(af1, bf, acc[1][ns]);
        }

        if (kt + 1 < nk) store_tile(cur ^ 1);
        __syncthreads();
    }

#pragma unroll
    for (int ms = 0; ms < 2; ++ms)
#pragma unroll
        for (int ns = 0; ns < 4; ++ns) {
            int gn = bn + wn * 64 + ns * 16 + ln;
            if (gn >= N) continue;
            float bval = bias ? bias[gn] : 0.f;
            int gm0 = bm + wm * 32 + ms * 16 + half * 8;
#pragma unroll
            for (int r = 0; r < 8; ++r) {
                int gm = gm0 + r;
                float v = acc[ms][ns][r] + bval;
                if (res) v += res[(size_t)gm * N + gn];
                if (relu) v = fmaxf(v, 0.f);
                C[(size_t)gm * N + gn] = v;
            }
        }
}

// ---------------------------------------------------------------------------
// RMSNorm rows (+optional exact GELU, +optional residual add)
// ---------------------------------------------------------------------------
__global__ void k_rmsnorm(const float* __restrict__ x, const float* __restrict__ w,
                          const float* __restrict__ res, float* __restrict__ y,
                          int C, int gelu)
{
    __shared__ float red[256];
    const int row = blockIdx.x, tid = threadIdx.x, bs = blockDim.x;
    const float* xr = x + (size_t)row * C;
    float s = 0.f;
    for (int c = tid; c < C; c += bs) { float v = xr[c]; s += v * v; }
    red[tid] = s; __syncthreads();
    for (int st = bs >> 1; st > 0; st >>= 1) {
        if (tid < st) red[tid] += red[tid + st];
        __syncthreads();
    }
    float scale = rsqrtf(red[0] / (float)C + 1.1920929e-07f);
    for (int c = tid; c < C; c += bs) {
        float v = xr[c] * scale * w[c];
        if (gelu) v = 0.5f * v * (1.f + erff(v * 0.70710678118654752f));
        if (res) v += res[(size_t)row * C + c];
        y[(size_t)row * C + c] = v;
    }
}

// ---------------------------------------------------------------------------
// QKV post: per-head LayerNorm(bias=False) + q scale + rotary; v LayerNorm.
// One block per token. qkv row layout: [8*128 q | 128 k | 192 v]
// ---------------------------------------------------------------------------
__global__ __launch_bounds__(256) void k_qkv_post(
    const float* __restrict__ qkv, const float* __restrict__ qw,
    const float* __restrict__ kw, const float* __restrict__ vw,
    float* __restrict__ outq, float* __restrict__ outk, float* __restrict__ outv)
{
    __shared__ float red[256];
    __shared__ float buf[128];
    const int n = blockIdx.x, tid = threadIdx.x;
    const float* row = qkv + (size_t)n * 1344;

    float cosv = 1.f, sinv = 0.f;
    if (tid < 128) {
        int c = tid & 63;
        float ex = 1.0f + (float)c * (8128.0f / 63.0f);    // linspace(1, 8129, 64)
        float invf = 1.0f / ((float)c + __powf(10.0f, ex)); // overflow -> inf -> 0
        float f = (float)n * invf;
        cosv = __cosf(f); sinv = __sinf(f);
    }

    for (int seg = 0; seg < 10; ++seg) {
        const int len = (seg < 9) ? 128 : 192;
        const int base = (seg < 8) ? seg * 128 : ((seg == 8) ? 1024 : 1152);
        float xv = (tid < len) ? row[base + tid] : 0.f;

        red[tid] = xv; __syncthreads();
        for (int st = 128; st > 0; st >>= 1) { if (tid < st) red[tid] += red[tid + st]; __syncthreads(); }
        float mu = red[0] / (float)len;
        __syncthreads();

        float d = (tid < len) ? (xv - mu) : 0.f;
        red[tid] = d * d; __syncthreads();
        for (int st = 128; st > 0; st >>= 1) { if (tid < st) red[tid] += red[tid + st]; __syncthreads(); }
        float var = red[0] / (float)len;
        __syncthreads();

        float yv = 0.f;
        if (tid < len) {
            const float* wp = (seg < 8) ? qw : ((seg == 8) ? kw : vw);
            yv = d * rsqrtf(var + 1e-5f) * wp[tid];
            if (seg < 8) yv *= 0.125f;   // SCALE = 64^-0.5
        }
        if (seg < 9) {
            if (tid < 128) buf[tid] = yv;
            __syncthreads();
            if (tid < 128) {
                float rot = (tid < 64) ? -buf[tid + 64] : buf[tid - 64];
                float o = buf[tid] * cosv + rot * sinv;
                if (seg < 8) outq[((size_t)seg * 2048 + n) * 128 + tid] = o;
                else         outk[(size_t)n * 128 + tid] = o;
            }
            __syncthreads();
        } else {
            if (tid < 192) outv[(size_t)n * 192 + tid] = yv;
            __syncthreads();
        }
    }
}

// ---------------------------------------------------------------------------
// Flash attention (streaming softmax), bf16 WMMA, dqk = 128 fixed.
// Block = 8 waves; each wave owns 16 query rows; 64 keys per iteration.
// Strides: Kls 68 uints (group stride 17), Vls 36 (stride 9), Pls 72 bf16 —
// all 16B-multiples, conflict-free or 2-way worst case for ds_load_b128.
// ---------------------------------------------------------------------------
template<int DV, bool BIAS, bool CLAMP, bool RESID>
__global__ __launch_bounds__(256) void k_flash(
    const float* __restrict__ Q, long long qBS, int qRS,
    const float* __restrict__ K, long long kBS, int kRS,
    const float* __restrict__ V, long long vBS, int vRS,
    float* __restrict__ O, long long oBS, int oRS,
    const float* __restrict__ R, long long rBS, int rRS,
    const float* __restrict__ bias, long long bBS, int bRow, int bCol,
    int seqK)
{
    constexpr int ND = DV / 16;
    __shared__ __attribute__((aligned(16))) unsigned int Kls[64][68];  // [key][dqk pair]
    __shared__ __attribute__((aligned(16))) unsigned int Vls[DV][36];  // [dim][key pair]
    __shared__ __attribute__((aligned(16))) __bf16 Pls[8][16][72];     // per-wave P

    const int b = blockIdx.y;
    const int tid = threadIdx.x;
    const int w = tid >> 5, lane = tid & 31, half = lane >> 4, ln = lane & 15;
    const int q0 = blockIdx.x * 128 + w * 16;

    // Load Q fragments once (fp32 -> bf16, A-layout swizzle)
    Frag aq[4];
    {
        const float* qp = Q + (size_t)b * qBS + (size_t)(q0 + ln) * qRS;
#pragma unroll
        for (int c = 0; c < 4; ++c)
#pragma unroll
            for (int j = 0; j < 8; ++j) {
                int kp = ((j & 4) << 1) + half * 4 + (j & 3);
                int kk = c * 32 + 2 * kp;
                aq[c].u[j] = pack2(qp[kk], qp[kk + 1]);
            }
    }

    const v8f vz = {0.f, 0.f, 0.f, 0.f, 0.f, 0.f, 0.f, 0.f};
    float m_i[8], l_i[8];
    v8f o_acc[ND];
#pragma unroll
    for (int r = 0; r < 8; ++r) { m_i[r] = -3.0e38f; l_i[r] = 0.f; }
#pragma unroll
    for (int d = 0; d < ND; ++d) o_acc[d] = vz;

    for (int kb = 0; kb < seqK; kb += 64) {
        // stage K tile
#pragma unroll
        for (int it = 0; it < 16; ++it) {
            int id = tid + 256 * it;
            int key = id >> 6, kp = id & 63;
            const float* p = K + (size_t)b * kBS + (size_t)(kb + key) * kRS + 2 * kp;
            Kls[key][kp] = pack2(p[0], p[1]);
        }
        // stage V tile transposed
        constexpr int VIT = (32 * DV) / 256;
#pragma unroll
        for (int it = 0; it < VIT; ++it) {
            int id = tid + 256 * it;
            int kp = id / DV, dim = id - kp * DV;
            const float* p = V + (size_t)b * vBS + (size_t)(kb + 2 * kp) * vRS + dim;
            Vls[dim][kp] = pack2(p[0], p[vRS]);
        }
        __syncthreads();

        // S = Q K^T (16 x 64 per wave)
        v8f s[4];
#pragma unroll
        for (int t = 0; t < 4; ++t) s[t] = vz;
#pragma unroll
        for (int c = 0; c < 4; ++c)
#pragma unroll
            for (int t = 0; t < 4; ++t) {
                v16bf bf = frag_b(&Kls[t * 16 + ln][0], c * 16);
                s[t] = wmma_bf16(aq[c].v, bf, s[t]);
            }

        // bias + softclamp
#pragma unroll
        for (int t = 0; t < 4; ++t)
#pragma unroll
            for (int r = 0; r < 8; ++r) {
                float x = s[t][r];
                if (BIAS) {
                    int qi = q0 + r + 8 * half;
                    int kj = kb + t * 16 + ln;
                    x += bias[(size_t)b * bBS + (size_t)(qi >> 3) * bRow + (size_t)(kj >> 3) * bCol];
                }
                if (CLAMP) x = 5.0f * tanhf(x * 0.2f);
                s[t][r] = x;
            }

        // online softmax (rows live in 16-lane halves -> xor reductions)
        float mnew[8], alpha[8], rsum[8];
#pragma unroll
        for (int r = 0; r < 8; ++r) {
            float mx = fmaxf(fmaxf(s[0][r], s[1][r]), fmaxf(s[2][r], s[3][r]));
#pragma unroll
            for (int msk = 1; msk < 16; msk <<= 1)
                mx = fmaxf(mx, __shfl_xor(mx, msk, 32));
            mnew[r] = fmaxf(m_i[r], mx);
            alpha[r] = __expf(m_i[r] - mnew[r]);
            rsum[r] = 0.f;
        }
#pragma unroll
        for (int t = 0; t < 4; ++t)
#pragma unroll
            for (int r = 0; r < 8; ++r) {
                float p = __expf(s[t][r] - mnew[r]);
                rsum[r] += p;
                Pls[w][r + 8 * half][t * 16 + ln] = (__bf16)p;
            }
#pragma unroll
        for (int r = 0; r < 8; ++r) {
#pragma unroll
            for (int msk = 1; msk < 16; msk <<= 1)
                rsum[r] += __shfl_xor(rsum[r], msk, 32);
            l_i[r] = l_i[r] * alpha[r] + rsum[r];
            m_i[r] = mnew[r];
        }
#pragma unroll
        for (int d = 0; d < ND; ++d)
#pragma unroll
            for (int r = 0; r < 8; ++r)
                o_acc[d][r] *= alpha[r];

        // cross-lane P store -> fragment reload within the wave
        asm volatile("s_wait_dscnt 0" ::: "memory");

        // O += P V
#pragma unroll
        for (int c = 0; c < 2; ++c) {
            v16bf pa = frag_a((const unsigned int*)&Pls[w][ln][0], c * 16);
#pragma unroll
            for (int d = 0; d < ND; ++d) {
                v16bf vb = frag_b(&Vls[d * 16 + ln][0], c * 16);
                o_acc[d] = wmma_bf16(pa, vb, o_acc[d]);
            }
        }
        __syncthreads();
    }

    float inv[8];
#pragma unroll
    for (int r = 0; r < 8; ++r) inv[r] = 1.0f / l_i[r];
#pragma unroll
    for (int d = 0; d < ND; ++d)
#pragma unroll
        for (int r = 0; r < 8; ++r) {
            int m = q0 + r + 8 * half;
            int col = d * 16 + ln;
            float val = o_acc[d][r] * inv[r];
            if (RESID) val += R[(size_t)b * rBS + (size_t)m * rRS + col];
            O[(size_t)b * oBS + (size_t)m * oRS + col] = val;
        }
}

// ---------------------------------------------------------------------------
// Host orchestration
// ---------------------------------------------------------------------------
extern "C" void kernel_launch(void* const* d_in, const int* in_sizes, int n_in,
                              void* d_out, int out_size, void* d_ws, size_t ws_size,
                              hipStream_t stream)
{
    (void)in_sizes; (void)n_in; (void)out_size; (void)ws_size;
    const float* single      = (const float*)d_in[0];
    const float* pairwise    = (const float*)d_in[1];
    const float* attn_pre_w  = (const float*)d_in[2];
    const float* attn_post_w = (const float*)d_in[3];
    const float* qkv_w       = (const float*)d_in[4];
    const float* q_norm_w    = (const float*)d_in[5];
    const float* k_norm_w    = (const float*)d_in[6];
    const float* v_norm_w    = (const float*)d_in[7];
    const float* bias_rms_w  = (const float*)d_in[8];
    const float* bias_proj_w = (const float*)d_in[9];
    const float* out_w       = (const float*)d_in[10];
    const float* ff_pre_w    = (const float*)d_in[11];
    const float* ff_post_w   = (const float*)d_in[12];
    const float* ff_w1       = (const float*)d_in[13];
    const float* ff_b1       = (const float*)d_in[14];
    const float* ff_w2       = (const float*)d_in[15];
    const float* ff_b2       = (const float*)d_in[16];
    const float* pw_attn_pre_w = (const float*)d_in[17];
    const float* pw_qk_w     = (const float*)d_in[18];
    const float* pw_v_w      = (const float*)d_in[19];
    const float* pw_v_b      = (const float*)d_in[20];
    const float* pw_ff_pre_w = (const float*)d_in[21];
    const float* pw_ff_w1    = (const float*)d_in[22];
    const float* pw_ff_b1    = (const float*)d_in[23];
    const float* pw_ff_w2    = (const float*)d_in[24];
    const float* pw_ff_b2    = (const float*)d_in[25];

    char* ws = (char*)d_ws;
    size_t off = 0;
    auto alloc = [&](size_t elems) -> float* {
        float* p = (float*)(ws + off);
        off += ((elems * sizeof(float)) + 255) & ~(size_t)255;
        return p;
    };
    float* sA    = alloc((size_t)2048 * 768);
    float* sB    = alloc((size_t)2048 * 768);
    float* xn    = alloc((size_t)2048 * 768);
    float* qkv   = alloc((size_t)2048 * 1344);
    float* bq    = alloc((size_t)8 * 2048 * 128);
    float* bk    = alloc((size_t)2048 * 128);
    float* bv    = alloc((size_t)2048 * 192);
    float* pbb   = alloc((size_t)65536 * 8);
    float* attno = alloc((size_t)2048 * 1536);
    float* proj  = alloc((size_t)2048 * 768);
    float* pn    = alloc((size_t)65536 * 128);
    float* pqk   = alloc((size_t)65536 * 256);
    float* pv    = alloc((size_t)65536 * 128);
    float* pA    = alloc((size_t)65536 * 128);
    float* pB    = alloc((size_t)65536 * 128);
    float* ffh   = attno;  // alias: attno consumed before ff1 writes
    float* pffh  = pqk;    // alias: pqk consumed by pairwise attn before pw-ff1

    const float* scur = single;
    const float* pcur = pairwise;
    for (int i = 0; i < 4; ++i) {
        // ---- attention block ----
        k_rmsnorm<<<2048, 256, 0, stream>>>(scur, attn_pre_w + i * 768, nullptr, xn, 768, 0);
        k_gemm<<<dim3(11, 16), 256, 0, stream>>>(xn, qkv_w + (size_t)i * 768 * 1344,
                                                 nullptr, nullptr, qkv, 2048, 1344, 768, 0);
        k_qkv_post<<<2048, 256, 0, stream>>>(qkv, q_norm_w + i * 128, k_norm_w + i * 128,
                                             v_norm_w + i * 192, bq, bk, bv);
        // pairwise bias path: gelu(rmsnorm(pairwise)) @ bias_proj
        k_rmsnorm<<<65536, 128, 0, stream>>>(pcur, bias_rms_w + i * 128, nullptr, pn, 128, 1);
        k_gemm<<<dim3(1, 512), 256, 0, stream>>>(pn, bias_proj_w + (size_t)i * 128 * 8,
                                                 nullptr, nullptr, pbb, 65536, 8, 128, 0);
        // flash attention, 8 heads, bias (nearest-repeat /8) + softclamp
        k_flash<192, true, true, false><<<dim3(16, 8), 256, 0, stream>>>(
            bq, 2048LL * 128, 128,
            bk, 0LL, 128,
            bv, 0LL, 192,
            attno, 192LL, 1536,
            nullptr, 0LL, 0,
            pbb, 1LL, 2048, 8,
            2048);
        k_gemm<<<dim3(6, 16), 256, 0, stream>>>(attno, out_w + (size_t)i * 1536 * 768,
                                                nullptr, nullptr, proj, 2048, 768, 1536, 0);
        k_rmsnorm<<<2048, 256, 0, stream>>>(proj, attn_post_w + i * 768, scur, sB, 768, 0);

        // ---- feedforward ----
        k_rmsnorm<<<2048, 256, 0, stream>>>(sB, ff_pre_w + i * 768, nullptr, xn, 768, 0);
        k_gemm<<<dim3(12, 16), 256, 0, stream>>>(xn, ff_w1 + (size_t)i * 768 * 1536,
                                                 ff_b1 + i * 1536, nullptr, ffh, 2048, 1536, 768, 1);
        k_gemm<<<dim3(6, 16), 256, 0, stream>>>(ffh, ff_w2 + (size_t)i * 1536 * 768,
                                                ff_b2 + i * 768, nullptr, proj, 2048, 768, 1536, 0);
        k_rmsnorm<<<2048, 256, 0, stream>>>(proj, ff_post_w + i * 768, sB, sA, 768, 0);
        scur = sA;

        // ---- pairwise track (every other layer) ----
        if ((i & 1) == 0) {
            int j = i >> 1;
            k_rmsnorm<<<65536, 128, 0, stream>>>(pcur, pw_attn_pre_w + j * 128, nullptr, pn, 128, 0);
            k_gemm<<<dim3(2, 512), 256, 0, stream>>>(pn, pw_qk_w + (size_t)j * 128 * 256,
                                                     nullptr, nullptr, pqk, 65536, 256, 128, 0);
            k_gemm<<<dim3(1, 512), 256, 0, stream>>>(pn, pw_v_w + (size_t)j * 128 * 128,
                                                     pw_v_b + j * 128, nullptr, pv, 65536, 128, 128, 0);
            // per-row attention over 256 keys, no scale/bias/clamp, +residual
            k_flash<128, false, false, true><<<dim3(2, 256), 256, 0, stream>>>(
                pqk, 256LL * 256, 256,
                pqk + 128, 256LL * 256, 256,
                pv, 256LL * 128, 128,
                pB, 256LL * 128, 128,
                pcur, 256LL * 128, 128,
                nullptr, 0LL, 0, 0,
                256);
            k_rmsnorm<<<65536, 128, 0, stream>>>(pB, pw_ff_pre_w + j * 128, nullptr, pn, 128, 0);
            k_gemm<<<dim3(2, 512), 256, 0, stream>>>(pn, pw_ff_w1 + (size_t)j * 128 * 256,
                                                     pw_ff_b1 + j * 256, nullptr, pffh, 65536, 256, 128, 1);
            k_gemm<<<dim3(1, 512), 256, 0, stream>>>(pffh, pw_ff_w2 + (size_t)j * 256 * 128,
                                                     pw_ff_b2 + j * 128, pB, pA, 65536, 128, 256, 0);
            pcur = pA;
        }
    }

    hipMemcpyAsync(d_out, scur, (size_t)2048 * 768 * sizeof(float),
                   hipMemcpyDeviceToDevice, stream);
    hipMemcpyAsync((char*)d_out + (size_t)2048 * 768 * sizeof(float), pcur,
                   (size_t)65536 * 128 * sizeof(float), hipMemcpyDeviceToDevice, stream);
}
